// PointNet2_52321291600075
// MI455X (gfx1250) — matrix operand order repaired and checked by reading, not compiled, not run on hardware
//
#include <hip/hip_runtime.h>
#include <hip/hip_bf16.h>
#include <math.h>

// ---------------------------------------------------------------------------
// MI455X (gfx1250, wave32) Point Transformer forward pass.
// Dense layers -> v_wmma_f32_16x16x32_f16 with vectorized LDS staging
// (b128 ds loads for fragments, float4 global loads for tiles).
// ---------------------------------------------------------------------------

typedef __attribute__((ext_vector_type(4)))  _Float16 v4h;
typedef __attribute__((ext_vector_type(8)))  _Float16 v8h;
typedef __attribute__((ext_vector_type(16))) _Float16 v16h;
typedef __attribute__((ext_vector_type(4)))  float    v4f;
typedef __attribute__((ext_vector_type(8)))  float    v8f;

#define WPB 8            // waves per block in the GEMM
#define KNN 16
#define NSTRIDE 17       // 16 neighbors + self loop
#define BN_EPS 1e-5f

__device__ __forceinline__ float relu_f(float x) { return fmaxf(x, 0.f); }

__device__ __forceinline__ v16h cat8(v8h lo, v8h hi) {
  return __builtin_shufflevector(lo, hi, 0, 1, 2, 3, 4, 5, 6, 7,
                                 8, 9, 10, 11, 12, 13, 14, 15);
}

// ---------------- WMMA GEMM: C[M,N] = act(A[M,K] @ B[K,N] + bias[N]) -------
// One wave computes a 32x16 C tile (two stacked 16x16 WMMA tiles sharing the
// B fragment). A tile staged row-major f16 (fragment = 2x ds_load_b128/lane);
// B tile staged K-transposed so a lane's fragment is two runs of 8 contiguous halves.
__global__ void __launch_bounds__(256) gemm_wmma_kernel(
    const float* __restrict__ A, const float* __restrict__ Bm,
    const float* __restrict__ bias, float* __restrict__ C,
    int M, int K, int N, int act)
{
  __shared__ _Float16 sA[WPB][32 * 32];    // 2 KB / wave
  __shared__ _Float16 sBt[WPB][16 * 32];   // 1 KB / wave (B transposed)
  const int wave  = threadIdx.x >> 5;
  const int lane  = threadIdx.x & 31;
  const int row0  = (blockIdx.y * WPB + wave) * 32;
  const int col0  = blockIdx.x * 16;
  const int mrow  = lane & 15;
  const int khalf = lane >> 4;
  _Float16* a_s = sA[wave];
  _Float16* b_s = sBt[wave];
  v8f acc0 = {0.f, 0.f, 0.f, 0.f, 0.f, 0.f, 0.f, 0.f};
  v8f acc1 = {0.f, 0.f, 0.f, 0.f, 0.f, 0.f, 0.f, 0.f};

  for (int k0 = 0; k0 < K; k0 += 32) {
    const bool fullA = (row0 + 32 <= M) && (k0 + 32 <= K);
    const bool fullB = (k0 + 32 <= K) && (col0 + 16 <= N);

    if (fullA) {
      if (k0 + 64 < K)
        __builtin_prefetch(&A[(long)row0 * K + k0 + 32], 0, 3);
#pragma unroll
      for (int p = 0; p < 8; ++p) {        // 1024 elems = 8 x 32 lanes x 4
        int idx = (p * 32 + lane) * 4;
        int r = idx >> 5, c = idx & 31;
        const v4f v = *(const v4f*)(&A[(long)(row0 + r) * K + k0 + c]);
        v4h h;
#pragma unroll
        for (int i = 0; i < 4; ++i) h[i] = (_Float16)v[i];
        *(v4h*)(&a_s[r * 32 + c]) = h;
      }
    } else {
      for (int t = lane; t < 1024; t += 32) {
        int r = t >> 5, c = t & 31;
        int gr = row0 + r, gc = k0 + c;
        a_s[t] = (_Float16)((gr < M && gc < K) ? A[(long)gr * K + gc] : 0.f);
      }
    }

    if (fullB) {
#pragma unroll
      for (int p = 0; p < 4; ++p) {        // 512 elems = 4 x 32 lanes x 4
        int t = p * 32 + lane;
        int r = t >> 2, c = (t & 3) * 4;   // K-row r, cols c..c+3
        const v4f v = *(const v4f*)(&Bm[(long)(k0 + r) * N + col0 + c]);
#pragma unroll
        for (int i = 0; i < 4; ++i) b_s[(c + i) * 32 + r] = (_Float16)v[i];
      }
    } else {
      for (int t = lane; t < 512; t += 32) {
        int r = t >> 4, c = t & 15;
        int gr = k0 + r, gc = col0 + c;
        b_s[c * 32 + r] = (_Float16)((gr < K && gc < N) ? Bm[(long)gr * N + gc] : 0.f);
      }
    }
    __syncthreads();

    // B fragment: lane holds col = mrow, K in [khalf*8,+8) and [16+khalf*8,+8).
    v16h bf = cat8(*(const v8h*)(&b_s[mrow * 32 + khalf * 8]),
                   *(const v8h*)(&b_s[mrow * 32 + 16 + khalf * 8]));
    // A fragments: lane holds row, K in [khalf*8,+8) and [16+khalf*8,+8).
    v16h af0 = cat8(*(const v8h*)(&a_s[mrow * 32 + khalf * 8]),
                    *(const v8h*)(&a_s[mrow * 32 + 16 + khalf * 8]));
    v16h af1 = cat8(*(const v8h*)(&a_s[(16 + mrow) * 32 + khalf * 8]),
                    *(const v8h*)(&a_s[(16 + mrow) * 32 + 16 + khalf * 8]));

    acc0 = __builtin_amdgcn_wmma_f32_16x16x32_f16(
        false, af0, false, bf, (short)0, acc0, false, false);
    acc1 = __builtin_amdgcn_wmma_f32_16x16x32_f16(
        false, af1, false, bf, (short)0, acc1, false, false);
    __syncthreads();
  }

  // C layout: VGPR r -> row (lane/16)*8 + r, col lane%16
  const int gc = col0 + mrow;
#pragma unroll
  for (int r = 0; r < 8; ++r) {
    int gr0 = row0 + khalf * 8 + r;
    int gr1 = gr0 + 16;
    float b = (bias && gc < N) ? bias[gc] : 0.f;
    if (gr0 < M && gc < N) {
      float v = acc0[r] + b;
      if (act) v = relu_f(v);
      C[(long)gr0 * N + gc] = v;
    }
    if (gr1 < M && gc < N) {
      float v = acc1[r] + b;
      if (act) v = relu_f(v);
      C[(long)gr1 * N + gc] = v;
    }
  }
}

// ---------------- initial embedding: rows are all-ones @ W[1,C] + b --------
__global__ void fill_embed_kernel(const float* __restrict__ W,
                                  const float* __restrict__ b,
                                  float* __restrict__ out, int R, int C)
{
  int t = blockIdx.x * blockDim.x + threadIdx.x;
  if (t >= R * C) return;
  int c = t % C;
  out[t] = W[c] + b[c];
}

// ---------------- BatchNorm (training stats over all rows) -----------------
__global__ void bn_stats_kernel(const float* __restrict__ x,
                                float* __restrict__ mean,
                                float* __restrict__ istd, int R, int C)
{
  __shared__ float s1[256], s2[256];
  int c = blockIdx.x;
  float a = 0.f, q = 0.f;
  for (int r = threadIdx.x; r < R; r += 256) {
    float v = x[(long)r * C + c];
    a += v; q += v * v;
  }
  s1[threadIdx.x] = a; s2[threadIdx.x] = q;
  __syncthreads();
  for (int o = 128; o > 0; o >>= 1) {
    if (threadIdx.x < o) {
      s1[threadIdx.x] += s1[threadIdx.x + o];
      s2[threadIdx.x] += s2[threadIdx.x + o];
    }
    __syncthreads();
  }
  if (threadIdx.x == 0) {
    float m = s1[0] / (float)R;
    float var = s2[0] / (float)R - m * m;
    mean[c] = m;
    istd[c] = rsqrtf(var + BN_EPS);
  }
}

__global__ void bn_apply_relu_kernel(const float* __restrict__ x,
                                     const float* __restrict__ mean,
                                     const float* __restrict__ istd,
                                     const float* __restrict__ g,
                                     const float* __restrict__ be,
                                     float* __restrict__ y, int R, int C)
{
  int t = blockIdx.x * blockDim.x + threadIdx.x;
  if (t >= R * C) return;
  int c = t % C;
  y[t] = relu_f((x[t] - mean[c]) * istd[c] * g[c] + be[c]);
}

// ---------------- brute-force kNN (LDS-tiled base points) ------------------
__global__ void knn_kernel(const float* __restrict__ q, int Nq,
                           const float* __restrict__ base, int Nb,
                           int* __restrict__ out, int stride,
                           int exclude_self, int append_self)
{
  __shared__ float sx[256], sy[256], sz[256];
  int t = blockIdx.x * 256 + threadIdx.x;
  float qx = 0.f, qy = 0.f, qz = 0.f;
  if (t < Nq) { qx = q[t * 3]; qy = q[t * 3 + 1]; qz = q[t * 3 + 2]; }
  float bd[KNN]; int bi[KNN];
#pragma unroll
  for (int k = 0; k < KNN; ++k) { bd[k] = 3.4e38f; bi[k] = 0; }
  for (int tile = 0; tile < Nb; tile += 256) {
    int n = min(256, Nb - tile);
    if ((int)threadIdx.x < n) {
      int g = tile + threadIdx.x;
      sx[threadIdx.x] = base[g * 3];
      sy[threadIdx.x] = base[g * 3 + 1];
      sz[threadIdx.x] = base[g * 3 + 2];
    }
    __syncthreads();
    if (t < Nq) {
      for (int j = 0; j < n; ++j) {
        int gi = tile + j;
        if (exclude_self && gi == t) continue;
        float dx = qx - sx[j], dy = qy - sy[j], dz = qz - sz[j];
        float d = dx * dx + dy * dy + dz * dz;
        if (d < bd[KNN - 1]) {
          int p = KNN - 1;
          while (p > 0 && bd[p - 1] > d) {
            bd[p] = bd[p - 1]; bi[p] = bi[p - 1]; --p;
          }
          bd[p] = d; bi[p] = gi;
        }
      }
    }
    __syncthreads();
  }
  if (t < Nq) {
#pragma unroll
    for (int k = 0; k < KNN; ++k) out[(long)t * stride + k] = bi[k];
    if (append_self) out[(long)t * stride + KNN] = t;
  }
}

// ---------------- farthest point sampling (one workgroup per batch) --------
__global__ void fps_kernel(const float* __restrict__ pos, int N, int nS,
                           int* __restrict__ out)
{
  __shared__ float dists[8192];
  __shared__ float rd[256];
  __shared__ int   ri[256];
  __shared__ int   s_last;
  pos += (long)blockIdx.x * N * 3;
  out += (long)blockIdx.x * nS;
  int tid = threadIdx.x;
  for (int i = tid; i < N; i += 256) dists[i] = 3.4e38f;
  if (tid == 0) { out[0] = 0; s_last = 0; }
  __syncthreads();
  for (int s = 1; s < nS; ++s) {
    float lx = pos[s_last * 3], ly = pos[s_last * 3 + 1], lz = pos[s_last * 3 + 2];
    float bestd = -1.f; int besti = 0;
    for (int i = tid; i < N; i += 256) {
      float dx = pos[i * 3] - lx, dy = pos[i * 3 + 1] - ly, dz = pos[i * 3 + 2] - lz;
      float d = fminf(dists[i], dx * dx + dy * dy + dz * dz);
      dists[i] = d;
      if (d > bestd) { bestd = d; besti = i; }
    }
    rd[tid] = bestd; ri[tid] = besti;
    __syncthreads();
    for (int o = 128; o > 0; o >>= 1) {
      if (tid < o) {
        if (rd[tid + o] > rd[tid] ||
            (rd[tid + o] == rd[tid] && ri[tid + o] < ri[tid])) {
          rd[tid] = rd[tid + o]; ri[tid] = ri[tid + o];
        }
      }
      __syncthreads();
    }
    if (tid == 0) { out[s] = ri[0]; s_last = ri[0]; }
    __syncthreads();
  }
}

// ---------------- gather sampled positions ---------------------------------
__global__ void gather_pos_kernel(const float* __restrict__ pos,
                                  const int* __restrict__ idx,
                                  float* __restrict__ sub, int B, int N, int nS)
{
  int t = blockIdx.x * blockDim.x + threadIdx.x;
  if (t >= B * nS) return;
  int b = t / nS;
  int g = b * N + idx[t];
  sub[t * 3]     = pos[g * 3];
  sub[t * 3 + 1] = pos[g * 3 + 1];
  sub[t * 3 + 2] = pos[g * 3 + 2];
}

// ---------------- per-edge position difference -----------------------------
__global__ void posdiff_kernel(const float* __restrict__ pos,
                               const int* __restrict__ nbr,
                               float* __restrict__ pd, int Bn, int N)
{
  int t = blockIdx.x * blockDim.x + threadIdx.x;
  if (t >= Bn * NSTRIDE) return;
  int row = t / NSTRIDE, j = t % NSTRIDE;
  int b = row / N;
  int g = b * N + nbr[(long)row * NSTRIDE + j];
  pd[(long)t * 3]     = pos[row * 3]     - pos[g * 3];
  pd[(long)t * 3 + 1] = pos[row * 3 + 1] - pos[g * 3 + 1];
  pd[(long)t * 3 + 2] = pos[row * 3 + 2] - pos[g * 3 + 2];
}

// ---------------- alpha input: a_dst[i] - a_src[nb] + delta ----------------
__global__ void alpha_in_kernel(const float* __restrict__ adst,
                                const float* __restrict__ asrc,
                                const float* __restrict__ delta,
                                const int* __restrict__ nbr,
                                float* __restrict__ out, int Bn, int N, int C)
{
  long t = (long)blockIdx.x * blockDim.x + threadIdx.x;
  if (t >= (long)Bn * NSTRIDE * C) return;
  long e = t / C; int c = t % C;
  int row = (int)(e / NSTRIDE), j = (int)(e % NSTRIDE);
  int b = row / N;
  int g = b * N + nbr[(long)row * NSTRIDE + j];
  out[t] = adst[(long)row * C + c] - asrc[(long)g * C + c] + delta[t];
}

// ---------------- softmax over neighbors + weighted aggregation ------------
__global__ void attn_pool_kernel(const float* __restrict__ alpha,
                                 const float* __restrict__ delta,
                                 const float* __restrict__ xl,
                                 const int* __restrict__ nbr,
                                 float* __restrict__ out, int Bn, int N, int C)
{
  long t = (long)blockIdx.x * blockDim.x + threadIdx.x;
  if (t >= (long)Bn * C) return;
  int row = (int)(t / C), c = (int)(t % C);
  int b = row / N;
  long ebase = (long)row * NSTRIDE;
  float mx = -3.4e38f;
  for (int j = 0; j < NSTRIDE; ++j)
    mx = fmaxf(mx, alpha[(ebase + j) * C + c]);
  float s = 0.f;
  for (int j = 0; j < NSTRIDE; ++j)
    s += expf(alpha[(ebase + j) * C + c] - mx);
  float acc = 0.f;
  for (int j = 0; j < NSTRIDE; ++j) {
    int g = b * N + nbr[ebase + j];
    float w = expf(alpha[(ebase + j) * C + c] - mx) / s;
    acc += w * (xl[(long)g * C + c] + delta[(ebase + j) * C + c]);
  }
  out[t] = acc;
}

// ---------------- k-max pooling over K=16 neighbors ------------------------
__global__ void pool_max_kernel(const float* __restrict__ h,
                                const int* __restrict__ nbr16,
                                float* __restrict__ xo, int BnS, int nS,
                                int N, int C)
{
  long t = (long)blockIdx.x * blockDim.x + threadIdx.x;
  if (t >= (long)BnS * C) return;
  int row = (int)(t / C), c = (int)(t % C);
  int b = row / nS;
  const int* nb = nbr16 + (long)row * KNN;
  float m = -3.4e38f;
#pragma unroll
  for (int k = 0; k < KNN; ++k)
    m = fmaxf(m, h[(long)(b * N + nb[k]) * C + c]);
  xo[t] = m;
}

// ---------------- output heads ---------------------------------------------
__global__ void feat_copy_kernel(const float* __restrict__ feat,
                                 float* __restrict__ out, long total)
{
  long t = (long)blockIdx.x * blockDim.x + threadIdx.x;
  if (t >= total) return;
  long p = t >> 7; int c = (int)(t & 127);
  out[p * 132 + c] = feat[t];
}

__global__ void head_kernel(const float* __restrict__ feat,
                            const float* __restrict__ offW,
                            const float* __restrict__ offb,
                            const float* __restrict__ clsW,
                            const float* __restrict__ clsb,
                            float* __restrict__ out, int P)
{
  int p = blockIdx.x * blockDim.x + threadIdx.x;
  if (p >= P) return;
  const float* f = feat + (long)p * 128;
  float o0 = 0.f, o1 = 0.f, o2 = 0.f, sc = 0.f;
  for (int c = 0; c < 128; ++c) {
    float v = f[c];
    o0 += v * offW[c * 3];
    o1 += v * offW[c * 3 + 1];
    o2 += v * offW[c * 3 + 2];
    sc += v * clsW[c];
  }
  float* orow = out + (long)p * 132;
  orow[128] = (o0 + offb[0]) * 0.1f;
  orow[129] = (o1 + offb[1]) * 0.1f;
  orow[130] = (o2 + offb[2]) * 0.1f;
  orow[131] = 1.f / (1.f + expf(-(sc + clsb[0])));
}

// ===========================================================================
struct TBP {
  const float *Win, *bin, *Wout, *bout, *Wsrc, *Wdst, *Wlin;
  const float *pW1, *pb1, *pW2, *pb2, *aW1, *ab1, *aW2, *ab2;
};

static inline size_t align256(size_t x) { return (x + 255) & ~(size_t)255; }
static inline int cdiv(long a, long b) { return (int)((a + b - 1) / b); }

extern "C" void kernel_launch(void* const* d_in, const int* in_sizes, int n_in,
                              void* d_out, int out_size, void* d_ws, size_t ws_size,
                              hipStream_t stream)
{
  (void)in_sizes; (void)out_size; (void)ws_size;
  if (n_in < 105) return;

  // ---- input pointers (setup_inputs dict order, params flattened in
  //      insertion order) ----
  const float* points = (const float*)d_in[0];   // [2, 8192, 3]
  // d_in[1] = vectors (unused by the reference forward pass)
  const float* in_W  = (const float*)d_in[2];
  const float* in_b  = (const float*)d_in[3];
  const float* in_g  = (const float*)d_in[4];
  const float* in_be = (const float*)d_in[5];
  const float* lin_W  = (const float*)d_in[6];   // [512, 32768]
  const float* lin_b  = (const float*)d_in[7];
  const float* lin_g  = (const float*)d_in[8];
  const float* lin_be = (const float*)d_in[9];
  const float* off_W = (const float*)d_in[10];
  const float* off_b = (const float*)d_in[11];
  const float* cls_W = (const float*)d_in[12];
  const float* cls_b = (const float*)d_in[13];

  int pi = 14;
  auto nextp = [&]() { return (const float*)d_in[pi++]; };
  TBP tbs[5];
  const float *tdW[4], *tdb[4], *tdg[4], *tdbe[4];
  auto read_tb = [&](TBP& t) {
    t.Win = nextp(); t.bin = nextp(); t.Wout = nextp(); t.bout = nextp();
    t.Wsrc = nextp(); t.Wdst = nextp(); t.Wlin = nextp();
    t.pW1 = nextp(); t.pb1 = nextp(); t.pW2 = nextp(); t.pb2 = nextp();
    t.aW1 = nextp(); t.ab1 = nextp(); t.aW2 = nextp(); t.ab2 = nextp();
  };
  read_tb(tbs[0]);
  for (int i = 0; i < 4; ++i) {
    tdW[i] = nextp(); tdb[i] = nextp(); tdg[i] = nextp(); tdbe[i] = nextp();
    read_tb(tbs[i + 1]);
  }

  // ---- workspace carve-out ----
  char* wp = (char*)d_ws;
  auto alloc_f = [&](size_t n) { float* p = (float*)wp; wp += align256(n * 4); return p; };
  auto alloc_i = [&](size_t n) { int* p = (int*)wp;   wp += align256(n * 4); return p; };

  const int B = 2;
  const int NL[5] = {8192, 2048, 512, 128, 32};
  const int CL[5] = {32, 64, 128, 256, 512};
  const long E0 = (long)B * 8192 * NSTRIDE;      // max edge count (level 0)

  float* f_tmp  = alloc_f(2097152);              // pre-BN buffer / final GEMM
  float* f_feat = alloc_f(2097152);              // [B, 8192, 128]
  float* f_h    = alloc_f(1048576);              // BN-relu'd transition features
  float* xA     = alloc_f(524288);               // x ping
  float* xB     = alloc_f(524288);               // x pong
  float* f_xin  = alloc_f(524288);
  float* f_asrc = alloc_f(524288);
  float* f_adst = alloc_f(524288);
  float* f_xl   = alloc_f(524288);
  float* pos_a  = alloc_f(12288);
  float* pos_b  = alloc_f(12288);
  float* s_mean = alloc_f(32768);
  float* s_istd = alloc_f(32768);
  float* e_pd   = alloc_f(E0 * 3);
  float* e_hid  = alloc_f(E0 * 64);
  float* e_del  = alloc_f(E0 * 32);              // C*E peaks at level 0
  float* e_alp  = alloc_f(E0 * 32);
  int* nbr    = alloc_i((long)B * 8192 * NSTRIDE);
  int* nbr16  = alloc_i((long)B * 2048 * KNN);
  int* fpsidx = alloc_i(B * 2048);

  auto gemm = [&](const float* A, const float* W, const float* bias, float* C,
                  long M, int K, int N, int act) {
    dim3 g(cdiv(N, 16), cdiv(M, 32 * WPB));      // 32 rows per wave now
    gemm_wmma_kernel<<<g, 256, 0, stream>>>(A, W, bias, C, (int)M, K, N, act);
  };
  auto launch1d = [&](long total) { return dim3(cdiv(total, 256)); };

  // ---- transformer block ----
  auto run_tb = [&](const TBP& tb, const float* x_in, float* x_out,
                    const float* pos, int N, int C) {
    long Bn = (long)B * N;
    long E = Bn * NSTRIDE;
    gemm(x_in,  tb.Win,  tb.bin, f_xin,  Bn, C, C, 1);
    gemm(f_xin, tb.Wsrc, nullptr, f_asrc, Bn, C, C, 0);
    gemm(f_xin, tb.Wdst, nullptr, f_adst, Bn, C, C, 0);
    gemm(f_xin, tb.Wlin, nullptr, f_xl,   Bn, C, C, 0);
    posdiff_kernel<<<launch1d(E), 256, 0, stream>>>(pos, nbr, e_pd, (int)Bn, N);
    gemm(e_pd,  tb.pW1, tb.pb1, e_hid, E, 3,  64, 1);
    gemm(e_hid, tb.pW2, tb.pb2, e_del, E, 64, C,  1);
    alpha_in_kernel<<<launch1d(E * C), 256, 0, stream>>>(
        f_adst, f_asrc, e_del, nbr, e_alp, (int)Bn, N, C);
    gemm(e_alp, tb.aW1, tb.ab1, e_hid, E, C,  64, 1);
    gemm(e_hid, tb.aW2, tb.ab2, e_alp, E, 64, C,  1);
    attn_pool_kernel<<<launch1d(Bn * C), 256, 0, stream>>>(
        e_alp, e_del, f_xl, nbr, f_xin, (int)Bn, N, C);
    gemm(f_xin, tb.Wout, tb.bout, x_out, Bn, C, C, 1);
  };

  // ================= forward pass =================
  // 1) input embedding: ones @ in_W + in_b, then global BN + ReLU
  long R0 = (long)B * NL[0];
  fill_embed_kernel<<<launch1d(R0 * 32), 256, 0, stream>>>(in_W, in_b, f_tmp, (int)R0, 32);
  bn_stats_kernel<<<32, 256, 0, stream>>>(f_tmp, s_mean, s_istd, (int)R0, 32);
  bn_apply_relu_kernel<<<launch1d(R0 * 32), 256, 0, stream>>>(
      f_tmp, s_mean, s_istd, in_g, in_be, xA, (int)R0, 32);

  // 2) level-0 knn graph (exclude self, append self loop) + tb0
  const float* pos_cur = points;
  for (int b = 0; b < B; ++b)
    knn_kernel<<<cdiv(NL[0], 256), 256, 0, stream>>>(
        pos_cur + (long)b * NL[0] * 3, NL[0], pos_cur + (long)b * NL[0] * 3,
        NL[0], nbr + (long)b * NL[0] * NSTRIDE, NSTRIDE, 1, 1);
  run_tb(tbs[0], xA, xB, pos_cur, NL[0], CL[0]);
  float* x_cur = xB;

  // 3) four transition-down + transformer stages
  for (int l = 0; l < 4; ++l) {
    int N = NL[l], N2 = NL[l + 1], C = CL[l], C2 = CL[l + 1];
    long R = (long)B * N, R2 = (long)B * N2;

    gemm(x_cur, tdW[l], tdb[l], f_tmp, R, C, C2, 0);
    bn_stats_kernel<<<C2, 256, 0, stream>>>(f_tmp, s_mean, s_istd, (int)R, C2);
    bn_apply_relu_kernel<<<launch1d(R * C2), 256, 0, stream>>>(
        f_tmp, s_mean, s_istd, tdg[l], tdbe[l], f_h, (int)R, C2);

    fps_kernel<<<B, 256, 0, stream>>>(pos_cur, N, N2, fpsidx);
    float* pos_next = (l & 1) ? pos_b : pos_a;
    gather_pos_kernel<<<launch1d(R2), 256, 0, stream>>>(
        pos_cur, fpsidx, pos_next, B, N, N2);

    for (int b = 0; b < B; ++b)
      knn_kernel<<<cdiv(N2, 256), 256, 0, stream>>>(
          pos_next + (long)b * N2 * 3, N2, pos_cur + (long)b * N * 3, N,
          nbr16 + (long)b * N2 * KNN, KNN, 0, 0);
    pool_max_kernel<<<launch1d(R2 * C2), 256, 0, stream>>>(
        f_h, nbr16, xA, (int)R2, N2, N, C2);

    pos_cur = pos_next;
    for (int b = 0; b < B; ++b)
      knn_kernel<<<cdiv(N2, 256), 256, 0, stream>>>(
          pos_cur + (long)b * N2 * 3, N2, pos_cur + (long)b * N2 * 3, N2,
          nbr + (long)b * N2 * NSTRIDE, NSTRIDE, 1, 1);
    run_tb(tbs[l + 1], xA, xB, pos_cur, N2, C2);
    x_cur = xB;
  }

  // 4) final MLP [512 -> 32768] + BN + ReLU, viewed as [B, 8192, 128]
  gemm(x_cur, lin_W, lin_b, f_tmp, (long)B * 32, 512, 32768, 0);
  bn_stats_kernel<<<32768, 256, 0, stream>>>(f_tmp, s_mean, s_istd, B * 32, 32768);
  bn_apply_relu_kernel<<<launch1d((long)B * 32 * 32768), 256, 0, stream>>>(
      f_tmp, s_mean, s_istd, lin_g, lin_be, f_feat, B * 32, 32768);

  // 5) heads + concat into d_out [B, 8192, 132]
  int P = B * 8192;
  feat_copy_kernel<<<launch1d((long)P * 128), 256, 0, stream>>>(
      f_feat, (float*)d_out, (long)P * 128);
  head_kernel<<<launch1d(P), 256, 0, stream>>>(
      f_feat, off_W, off_b, cls_W, cls_b, (float*)d_out, P);
}
